// GNN_RNN_70171175682619
// MI455X (gfx1250) — matrix operand order repaired and checked by reading, not compiled, not run on hardware
//
#include <hip/hip_runtime.h>

// ---------------------------------------------------------------------------
// EdgeGATConv x2 (DGL semantics) for MI455X (gfx1250, wave32, WMMA).
// Strip-GEMM: one wave computes a 16x128 strip of C with 8 WMMA accumulators;
// B staged once per block into LDS as transposed f16. fe-GEMM fuses the
// ee = sum(fe*ae) attention dot as a register epilogue. Softmax + aggregation
// via global float atomics.
// ---------------------------------------------------------------------------

#define GN     50000      // nodes
#define GE     800000     // edges
#define GS     2          // snapshots
#define NDIM   128
#define EDIM   64
#define HEADS  4
#define DOUT   32
#define HD     128        // HEADS*DOUT
#define NEG_SLOPE 0.2f

typedef __attribute__((ext_vector_type(16))) _Float16 v16h;
typedef __attribute__((ext_vector_type(8)))  _Float16 v8h;
typedef __attribute__((ext_vector_type(8)))  float    v8f;

// ---------------------------------------------------------------------------
// C[M,128] = A[M,K] x B[K,128]  (row-major f32, f16 multiply / f32 accumulate)
// One wave32 computes one 16-row strip across all 8 column tiles.
// K % 32 == 0, M % 16 == 0. B is staged to LDS transposed (f16, padded).
// If aeFlat != nullptr also emits ee[row,h] = sum_d C[row,h*32+d]*aeFlat[h*32+d].
// ---------------------------------------------------------------------------
__global__ void __launch_bounds__(256)
wmma_gemm_strip(const float* __restrict__ A, const float* __restrict__ B,
                float* __restrict__ C, int K, int strips,
                const float* __restrict__ aeFlat, float* __restrict__ ee)
{
    __shared__ __align__(16) _Float16 Bt[128 * 136];   // [n][k], stride KP
    const int tid = (int)threadIdx.x;
    const int KP  = K + 8;                             // pad: keeps 16B align, kills conflicts

    // ---- stage B^T as f16 into LDS (coalesced float4 global reads) --------
    for (int idx = tid; idx < (K << 5); idx += 256) {  // K*128/4 float4s
        int k  = idx >> 5;
        int n4 = (idx & 31) << 2;
        float4 v = ((const float4*)B)[idx];
        Bt[(n4 + 0) * KP + k] = (_Float16)v.x;
        Bt[(n4 + 1) * KP + k] = (_Float16)v.y;
        Bt[(n4 + 2) * KP + k] = (_Float16)v.z;
        Bt[(n4 + 3) * KP + k] = (_Float16)v.w;
    }
    __syncthreads();

    const int wave  = tid >> 5;
    const int lane  = tid & 31;
    const int strip = (int)blockIdx.x * 8 + wave;
    if (strip >= strips) return;                       // wave-uniform

    const int half = lane >> 4;                        // 0: lanes 0-15, 1: 16-31
    const int l15  = lane & 15;
    const int rowA = strip * 16 + l15;

    v8f acc[8];
#pragma unroll
    for (int t = 0; t < 8; ++t) acc[t] = (v8f){};

    for (int k0 = 0; k0 < K; k0 += 32) {
        // A 16x32 f16 fragment: per lane two contiguous 8-float runs,
        // K-offsets [half*8, half*8+8) and [16+half*8, 16+half*8+8)
        const float4* Ap = (const float4*)(A + (size_t)rowA * K + k0 + (half << 3));
        float4 f0 = Ap[0], f1 = Ap[1], f2 = Ap[4], f3 = Ap[5];
        v16h a;
        a[0]  = (_Float16)f0.x; a[1]  = (_Float16)f0.y;
        a[2]  = (_Float16)f0.z; a[3]  = (_Float16)f0.w;
        a[4]  = (_Float16)f1.x; a[5]  = (_Float16)f1.y;
        a[6]  = (_Float16)f1.z; a[7]  = (_Float16)f1.w;
        a[8]  = (_Float16)f2.x; a[9]  = (_Float16)f2.y;
        a[10] = (_Float16)f2.z; a[11] = (_Float16)f2.w;
        a[12] = (_Float16)f3.x; a[13] = (_Float16)f3.y;
        a[14] = (_Float16)f3.z; a[15] = (_Float16)f3.w;

#pragma unroll
        for (int t = 0; t < 8; ++t) {
            // B 32x16 fragment: lane wants B[k0+half*8 + i][col] and
            // B[k0+16+half*8 + i][col], i=0..7, from Bt[col][...]
            const _Float16* bp = &Bt[(size_t)(t * 16 + l15) * KP + k0 + (half << 3)];
            v8h b0 = *(const v8h*)bp;
            v8h b1 = *(const v8h*)(bp + 16);
            v16h b;
#pragma unroll
            for (int i = 0; i < 8; ++i) { b[i] = b0[i]; b[8 + i] = b1[i]; }
            acc[t] = __builtin_amdgcn_wmma_f32_16x16x32_f16(
                         false, a, false, b, (short)0, acc[t], false, false);
        }
    }

    // ---- store C (C/D layout: VGPR r -> row r + half*8) -------------------
#pragma unroll
    for (int t = 0; t < 8; ++t) {
        int col = t * 16 + l15;
#pragma unroll
        for (int r = 0; r < 8; ++r)
            C[(size_t)(strip * 16 + half * 8 + r) * HD + col] = acc[t][r];
    }

    // ---- fused ee epilogue ------------------------------------------------
    if (aeFlat) {
        float hs[HEADS][8];
#pragma unroll
        for (int h = 0; h < HEADS; ++h)
#pragma unroll
            for (int r = 0; r < 8; ++r) hs[h][r] = 0.0f;

#pragma unroll
        for (int t = 0; t < 8; ++t) {
            float w = aeFlat[t * 16 + l15];            // ae flat over 128 cols
#pragma unroll
            for (int r = 0; r < 8; ++r) hs[t >> 1][r] += acc[t][r] * w;
        }
        // reduce over the 16 lanes of each half (xor masks stay below bit 4)
#pragma unroll
        for (int h = 0; h < HEADS; ++h)
#pragma unroll
            for (int r = 0; r < 8; ++r) {
                float v = hs[h][r];
                v += __shfl_xor(v, 1);
                v += __shfl_xor(v, 2);
                v += __shfl_xor(v, 4);
                v += __shfl_xor(v, 8);
                hs[h][r] = v;
            }
        if (l15 == 0) {
#pragma unroll
            for (int r = 0; r < 8; ++r) {
                int row = strip * 16 + half * 8 + r;
#pragma unroll
                for (int h = 0; h < HEADS; ++h)
                    ee[row * HEADS + h] = hs[h][r];
            }
        }
    }
}

// ---------------------------------------------------------------------------
__global__ void init_state(float* __restrict__ rst, float* __restrict__ m,
                           float* __restrict__ denom)
{
    int i = (int)(blockIdx.x * blockDim.x + threadIdx.x);
    if (i < GN * HD) rst[i] = 0.0f;
    if (i < GN * HEADS) { m[i] = -__builtin_huge_valf(); denom[i] = 0.0f; }
}

// el[n,h] = sum_d ft[n,h*32+d]*al[h,d]; er likewise with ar
__global__ void node_attn(const float* __restrict__ ft,
                          const float* __restrict__ al,
                          const float* __restrict__ ar,
                          float* __restrict__ el, float* __restrict__ er)
{
    int i = (int)(blockIdx.x * blockDim.x + threadIdx.x);
    if (i >= GN * HEADS) return;
    int n = i >> 2, h = i & 3;
    const float* f  = ft + (size_t)n * HD + h * DOUT;
    const float* Al = al + h * DOUT;
    const float* Ar = ar + h * DOUT;
    float sl = 0.0f, sr = 0.0f;
#pragma unroll 8
    for (int d = 0; d < DOUT; ++d) { float v = f[d]; sl += v * Al[d]; sr += v * Ar[d]; }
    el[i] = sl; er[i] = sr;
}

// ordered-int trick: valid with m initialized to -inf
__device__ __forceinline__ void atomicMaxF(float* addr, float v)
{
    if (v >= 0.0f) atomicMax((int*)addr, __float_as_int(v));
    else           atomicMin((unsigned int*)addr, __float_as_uint(v));
}

// logits = leaky(el[src]+er[dst]+ee); store; segment-max into m[dst]
__global__ void edge_logits(const float* __restrict__ ee,
                            const float* __restrict__ el,
                            const float* __restrict__ er,
                            const int* __restrict__ src,
                            const int* __restrict__ dst,
                            float* __restrict__ lg, float* __restrict__ m)
{
    int i = (int)(blockIdx.x * blockDim.x + threadIdx.x);
    if (i >= GE * HEADS) return;
    int e = i >> 2, h = i & 3;
    int s = src[e], dn = dst[e];
    float lo = el[s * HEADS + h] + er[dn * HEADS + h] + ee[i];
    lo = (lo >= 0.0f) ? lo : NEG_SLOPE * lo;
    lg[i] = lo;
    atomicMaxF(&m[dn * HEADS + h], lo);
}

// nodes with zero in-degree: m := 0
__global__ void m_fixup(float* __restrict__ m)
{
    int i = (int)(blockIdx.x * blockDim.x + threadIdx.x);
    if (i >= GN * HEADS) return;
    if (m[i] == -__builtin_huge_valf()) m[i] = 0.0f;
}

// ex = exp(logit - m[dst]); accumulate denom[dst]; lg overwritten with ex
__global__ void edge_exp(const int* __restrict__ dst,
                         const float* __restrict__ m,
                         float* __restrict__ lg, float* __restrict__ denom)
{
    int i = (int)(blockIdx.x * blockDim.x + threadIdx.x);
    if (i >= GE * HEADS) return;
    int e = i >> 2, h = i & 3;
    int dn = dst[e];
    float ex = expf(lg[i] - m[dn * HEADS + h]);
    lg[i] = ex;
    atomicAdd(&denom[dn * HEADS + h], ex);
}

// one wave per edge: rst[dst, :] += alpha_h * (ft[src, :] + fe[e, :])
__global__ void __launch_bounds__(256)
edge_scatter(const float* __restrict__ ft, const float* __restrict__ fe,
             const float* __restrict__ lg, const float* __restrict__ denom,
             const int* __restrict__ src, const int* __restrict__ dst,
             float* __restrict__ rst)
{
    int e = (int)((blockIdx.x * blockDim.x + threadIdx.x) >> 5);
    if (e >= GE) return;
    int lane = (int)(threadIdx.x & 31);
    int h = lane >> 3;                                 // j = lane*4 .. lane*4+3
    int s = src[e], dn = dst[e];
    float alpha = lg[e * HEADS + h] / denom[dn * HEADS + h];
    float4 a4 = ((const float4*)(ft + (size_t)s * HD))[lane];
    float4 b4 = ((const float4*)(fe + (size_t)e * HD))[lane];
    float* rp = rst + (size_t)dn * HD + lane * 4;
    atomicAdd(rp + 0, alpha * (a4.x + b4.x));
    atomicAdd(rp + 1, alpha * (a4.y + b4.y));
    atomicAdd(rp + 2, alpha * (a4.z + b4.z));
    atomicAdd(rp + 3, alpha * (a4.w + b4.w));
}

// out[n,d] = mean_h( rst[n,h,d] + res[n,h,d] + b[h,d] )
__global__ void finalize(const float* __restrict__ rst,
                         const float* __restrict__ res,
                         const float* __restrict__ b,
                         float* __restrict__ out)
{
    int i = (int)(blockIdx.x * blockDim.x + threadIdx.x);
    if (i >= GN * DOUT) return;
    int n = i >> 5, d = i & 31;
    float s = 0.0f;
#pragma unroll
    for (int h = 0; h < HEADS; ++h) {
        int j = h * DOUT + d;
        s += rst[(size_t)n * HD + j] + res[(size_t)n * HD + j] + b[j];
    }
    out[i] = 0.25f * s;
}

// ---------------------------------------------------------------------------
static inline int cdiv(long long a, int b) { return (int)((a + b - 1) / b); }

extern "C" void kernel_launch(void* const* d_in, const int* in_sizes, int n_in,
                              void* d_out, int out_size, void* d_ws, size_t ws_size,
                              hipStream_t stream)
{
    (void)in_sizes; (void)n_in; (void)out_size; (void)ws_size;

    const float* node_feats = (const float*)d_in[0];   // [S,N,128]
    const float* edge_feats = (const float*)d_in[1];   // [S,E,64]
    const int*   src_all    = (const int*)d_in[2];     // [S,E]
    const int*   dst_all    = (const int*)d_in[3];     // [S,E]
    const float* W[2]    = { (const float*)d_in[4],  (const float*)d_in[11] };
    const float* We[2]   = { (const float*)d_in[5],  (const float*)d_in[12] };
    const float* al[2]   = { (const float*)d_in[6],  (const float*)d_in[13] };
    const float* ar[2]   = { (const float*)d_in[7],  (const float*)d_in[14] };
    const float* ae[2]   = { (const float*)d_in[8],  (const float*)d_in[15] };
    const float* Wres[2] = { (const float*)d_in[9],  (const float*)d_in[16] };
    const float* bia[2]  = { (const float*)d_in[10], (const float*)d_in[17] };

    // workspace layout (floats)
    float* ws    = (float*)d_ws;
    float* ft    = ws;                 ws += (size_t)GN * HD;     // [N,128]
    float* res   = ws;                 ws += (size_t)GN * HD;     // [N,128]
    float* rst   = ws;                 ws += (size_t)GN * HD;     // [N,128]
    float* x1    = ws;                 ws += (size_t)GN * DOUT;   // [N,32]
    float* el    = ws;                 ws += (size_t)GN * HEADS;
    float* er    = ws;                 ws += (size_t)GN * HEADS;
    float* m     = ws;                 ws += (size_t)GN * HEADS;
    float* denom = ws;                 ws += (size_t)GN * HEADS;
    float* lg    = ws;                 ws += (size_t)GE * HEADS;  // logits -> ex
    float* eeb   = ws;                 ws += (size_t)GE * HEADS;  // edge attn dot
    float* fe    = ws;                 ws += (size_t)GE * HD;     // [E,128]

    const int TB = 256;

    for (int s = 0; s < GS; ++s) {
        const float* x0  = node_feats + (size_t)s * GN * NDIM;
        const float* eF  = edge_feats + (size_t)s * GE * EDIM;
        const int*   src = src_all + (size_t)s * GE;
        const int*   dst = dst_all + (size_t)s * GE;

        const float* xin = x0;
        int Kin = NDIM;                       // 128 for layer 0, 32 for layer 1

        for (int layer = 0; layer < 2; ++layer) {
            init_state<<<cdiv((long long)GN * HD, TB), TB, 0, stream>>>(rst, m, denom);

            // ft = xin @ W        [N,Kin] x [Kin,128]
            {
                int strips = GN / 16;
                wmma_gemm_strip<<<cdiv(strips, 8), TB, 0, stream>>>(
                    xin, W[layer], ft, Kin, strips, nullptr, nullptr);
            }
            // res = xin @ Wres
            {
                int strips = GN / 16;
                wmma_gemm_strip<<<cdiv(strips, 8), TB, 0, stream>>>(
                    xin, Wres[layer], res, Kin, strips, nullptr, nullptr);
            }
            // fe = eF @ We        [E,64] x [64,128], fused ee = sum(fe*ae)
            {
                int strips = GE / 16;
                wmma_gemm_strip<<<cdiv(strips, 8), TB, 0, stream>>>(
                    eF, We[layer], fe, EDIM, strips, ae[layer], eeb);
            }

            node_attn<<<cdiv((long long)GN * HEADS, TB), TB, 0, stream>>>(
                ft, al[layer], ar[layer], el, er);
            edge_logits<<<cdiv((long long)GE * HEADS, TB), TB, 0, stream>>>(
                eeb, el, er, src, dst, lg, m);
            m_fixup<<<cdiv((long long)GN * HEADS, TB), TB, 0, stream>>>(m);
            edge_exp<<<cdiv((long long)GE * HEADS, TB), TB, 0, stream>>>(
                dst, m, lg, denom);

            edge_scatter<<<cdiv((long long)GE * 32, TB), TB, 0, stream>>>(
                ft, fe, lg, denom, src, dst, rst);

            float* outp = (layer == 0) ? x1
                                       : ((float*)d_out + (size_t)s * GN * DOUT);
            finalize<<<cdiv((long long)GN * DOUT, TB), TB, 0, stream>>>(
                rst, res, bia[layer], outp);

            xin = x1;
            Kin = DOUT;
        }
    }
}
